// SparseAttentionBase_2052994367912
// MI455X (gfx1250) — matrix operand order repaired and checked by reading, not compiled, not run on hardware
//
#include <hip/hip_runtime.h>
#include <hip/hip_bf16.h>

// ---------------------------------------------------------------------------
// Fused QKV projection (bf16 WMMA GEMM, f32 accumulate) + RoPE + transpose.
//   X: (16384, 4096) f32   W = [Wq;Wk;Wv]: (6144, 4096) f32
//   out = concat(q(4,32,4096,128), k(4,8,4096,128), v(4,8,4096,128)) f32
// Pass 1 converts X and W to bf16 in d_ws (176 MB).
// Pass 2: 128x128 block tile, 8 waves (2x4), 64x32 per wave, K-step 32,
//         double-buffered LDS fed by GLOBAL_LOAD_ASYNC_TO_LDS_B128 (ASYNCcnt),
//         v_wmma_f32_16x16x32_bf16 inner loop, RoPE epilogue in registers.
// ---------------------------------------------------------------------------

typedef __attribute__((ext_vector_type(16))) __bf16 v16bf;
typedef __attribute__((ext_vector_type(8)))  __bf16 v8bf;
typedef __attribute__((ext_vector_type(8)))  float  v8f;

union ABFrag { v16bf f; v8bf h[2]; };

#define KDIM   4096
#define MTOT   16384            // BS*SEQ
#define NTOT   6144             // 4096 q + 1024 k + 1024 v
#define SEQLEN 4096
#define HDIM   128

// ----------------------------- pack kernel ---------------------------------
__global__ __launch_bounds__(256) void pack_bf16_kernel(
    const float* __restrict__ x,  const float* __restrict__ wq,
    const float* __restrict__ wk, const float* __restrict__ wv,
    __bf16* __restrict__ xb, __bf16* __restrict__ wb) {
  const size_t XTOT = (size_t)MTOT * KDIM;          // 67,108,864
  const size_t WQ   = (size_t)4096 * KDIM;          // 16,777,216
  const size_t WK   = (size_t)1024 * KDIM;          //  4,194,304
  size_t base = ((size_t)blockIdx.x * blockDim.x + threadIdx.x) * 8;

  const float* src;
  __bf16* dst;
  if (base < XTOT) {
    src = x + base;  dst = xb + base;
  } else {
    size_t wo = base - XTOT;
    dst = wb + wo;
    if      (wo < WQ)      src = wq + wo;
    else if (wo < WQ + WK) src = wk + (wo - WQ);
    else                   src = wv + (wo - WQ - WK);
  }
  float4 a = *(const float4*)(src);
  float4 b = *(const float4*)(src + 4);
  v8bf o;
  o[0] = (__bf16)a.x; o[1] = (__bf16)a.y; o[2] = (__bf16)a.z; o[3] = (__bf16)a.w;
  o[4] = (__bf16)b.x; o[5] = (__bf16)b.y; o[6] = (__bf16)b.z; o[7] = (__bf16)b.w;
  *(v8bf*)dst = o;
}

// ----------------------------- GEMM + RoPE ---------------------------------
// LDS row stride: 32 bf16 of data padded to 40 bf16 (80 B) -> the per-lane
// 16-B fragment reads land on distinct bank groups (stride 20 banks, period 16).
#define LDS_ROW 40
#define BUF_BYTES (128 * LDS_ROW * 2)     // 10240 B per matrix per buffer

// Async cache->LDS copy of 16 B (no VGPR round trip, tracked by ASYNCcnt).
__device__ __forceinline__ void async_copy16(uint32_t lds_off,
                                             const __bf16* gptr) {
  asm volatile("global_load_async_to_lds_b128 %0, %1, off"
               :: "v"(lds_off), "v"(gptr)
               : "memory");
}
__device__ __forceinline__ void wait_async0() {
  asm volatile("s_wait_asynccnt 0" ::: "memory");
}

__global__ __launch_bounds__(256) void qkv_wmma_rope_kernel(
    const __bf16* __restrict__ xb, const __bf16* __restrict__ wb,
    const float* __restrict__ cosT, const float* __restrict__ sinT,
    float* __restrict__ out) {
  __shared__ __attribute__((aligned(16))) __bf16 ldsA[2][128 * LDS_ROW];
  __shared__ __attribute__((aligned(16))) __bf16 ldsB[2][128 * LDS_ROW];

  const int t    = threadIdx.x;
  const int lane = t & 31;
  const int wid  = t >> 5;
  const int l15  = lane & 15;
  const bool hi  = lane >= 16;

  const int waveM  = (wid >> 2) * 64;   // 2 wave-rows of 64
  const int waveN  = (wid & 3) * 32;    // 4 wave-cols of 32
  const int blockN = blockIdx.x * 128;  // 48 blocks
  const int blockM = blockIdx.y * 128;  // 128 blocks

  // Staging: 512 16-B chunks per 128x32 tile (128 rows x 4 chunks), 2/thread.
  const int row0 = t >> 2;              // 0..63
  const int ch   = t & 3;               // 0..3
  const __bf16* gA0 = xb + (size_t)(blockM + row0) * KDIM + ch * 8;
  const __bf16* gA1 = gA0 + (size_t)64 * KDIM;
  const __bf16* gB0 = wb + (size_t)(blockN + row0) * KDIM + ch * 8;
  const __bf16* gB1 = gB0 + (size_t)64 * KDIM;

  // LDS aperture: low 32 bits of the flat shared pointer are the LDS byte
  // address the async-DMA instruction expects in its VDST VGPR.
  const uint32_t ldsA0 = (uint32_t)(uintptr_t)&ldsA[0][0];
  const uint32_t ldsB0 = (uint32_t)(uintptr_t)&ldsB[0][0];
  const uint32_t dChunk  = (uint32_t)(row0 * 80 + ch * 16);
  const uint32_t dChunk2 = (uint32_t)((row0 + 64) * 80 + ch * 16);

  auto issue_async = [&](int buf, int ko) {
    const uint32_t bA = ldsA0 + (uint32_t)buf * BUF_BYTES;
    const uint32_t bB = ldsB0 + (uint32_t)buf * BUF_BYTES;
    async_copy16(bA + dChunk,  gA0 + ko);
    async_copy16(bA + dChunk2, gA1 + ko);
    async_copy16(bB + dChunk,  gB0 + ko);
    async_copy16(bB + dChunk2, gB1 + ko);
  };

  v8f acc[4][2];
  const v8f zacc = {0.f, 0.f, 0.f, 0.f, 0.f, 0.f, 0.f, 0.f};
#pragma unroll
  for (int mi = 0; mi < 4; ++mi)
#pragma unroll
    for (int ni = 0; ni < 2; ++ni) acc[mi][ni] = zacc;

  issue_async(0, 0);
  wait_async0();
  __syncthreads();

  const int NKB = KDIM / 32;        // 128
  for (int kb = 0; kb < NKB; ++kb) {
    const int cur = kb & 1;

    // Kick off the DMA for the next K-step; it runs under this step's WMMAs.
    if (kb + 1 < NKB) issue_async(cur ^ 1, (kb + 1) * 32);

    const char* pA = (const char*)&ldsA[cur][0];
    const char* pB = (const char*)&ldsB[cur][0];

    // A fragment (16x32 bf16): lanes<16 take K[0:8)+K[16:24); lanes>=16
    // take K[8:16)+K[24:32) of row M=l15 (per ISA VGPR layout).
    ABFrag afr[4];
#pragma unroll
    for (int mi = 0; mi < 4; ++mi) {
      const char* p = pA + (waveM + mi * 16 + l15) * 80;
      afr[mi].h[0] = *(const v8bf*)(p + (hi ? 16 : 0));
      afr[mi].h[1] = *(const v8bf*)(p + (hi ? 48 : 32));
    }
    // B fragment (32x16 bf16): column N=l15; lanes<16 K[0:8)+K[16:24), lanes>=16 K[8:16)+K[24:32).
    ABFrag bfr[2];
#pragma unroll
    for (int ni = 0; ni < 2; ++ni) {
      const char* p = pB + (waveN + ni * 16 + l15) * 80;
      bfr[ni].h[0] = *(const v8bf*)(p + (hi ? 16 : 0));
      bfr[ni].h[1] = *(const v8bf*)(p + (hi ? 48 : 32));
    }

#pragma unroll
    for (int mi = 0; mi < 4; ++mi)
#pragma unroll
      for (int ni = 0; ni < 2; ++ni)
        acc[mi][ni] = __builtin_amdgcn_wmma_f32_16x16x32_bf16(
            false, afr[mi].f, false, bfr[ni].f, (short)0, acc[mi][ni],
            false, false);

    if (kb + 1 < NKB) {
      wait_async0();                // own wave's DMA for kb+1 has landed
      __syncthreads();              // all waves' DMA landed / reads done
    }
  }

  // ---------------- RoPE epilogue + transposed store ----------------------
  // C/D layout: VGPR r -> (M = r + (hi?8:0), N = l15). Adjacent head-dims sit
  // in adjacent lanes, so the rotation partner is shfl_xor(...,1).
  const size_t QSZ = (size_t)4 * 32 * SEQLEN * HDIM;  // 67,108,864
  const size_t KSZ = (size_t)4 * 8 * SEQLEN * HDIM;   // 16,777,216
#pragma unroll
  for (int mi = 0; mi < 4; ++mi) {
#pragma unroll
    for (int ni = 0; ni < 2; ++ni) {
      const int n = blockN + waveN + ni * 16 + l15;   // global out column
#pragma unroll
      for (int r = 0; r < 8; ++r) {
        const int m = blockM + waveM + mi * 16 + r + (hi ? 8 : 0);
        const int b = m >> 12;
        const int s = m & (SEQLEN - 1);
        const float val     = acc[mi][ni][r];
        const float partner = __shfl_xor(val, 1, 32);
        float res;
        size_t idx;
        if (n < 4096) {                               // Q head, apply RoPE
          const int h = n >> 7, d = n & 127, i = d >> 1;
          const float c  = cosT[s * 64 + i];
          const float sn = sinT[s * 64 + i];
          res = (d & 1) ? fmaf(partner, sn, val * c)
                        : fmaf(-partner, sn, val * c);
          idx = ((((size_t)b * 32 + h) * SEQLEN + (size_t)s) << 7) + d;
        } else if (n < 5120) {                        // K head, apply RoPE
          const int nk = n - 4096;
          const int h = nk >> 7, d = nk & 127, i = d >> 1;
          const float c  = cosT[s * 64 + i];
          const float sn = sinT[s * 64 + i];
          res = (d & 1) ? fmaf(partner, sn, val * c)
                        : fmaf(-partner, sn, val * c);
          idx = QSZ + ((((size_t)b * 8 + h) * SEQLEN + (size_t)s) << 7) + d;
        } else {                                      // V head, passthrough
          const int nv = n - 5120;
          const int h = nv >> 7, d = nv & 127;
          res = val;
          idx = QSZ + KSZ +
                ((((size_t)b * 8 + h) * SEQLEN + (size_t)s) << 7) + d;
        }
        out[idx] = res;
      }
    }
  }
}

// ----------------------------- launcher ------------------------------------
extern "C" void kernel_launch(void* const* d_in, const int* in_sizes, int n_in,
                              void* d_out, int out_size, void* d_ws, size_t ws_size,
                              hipStream_t stream) {
  (void)in_sizes; (void)n_in; (void)out_size; (void)ws_size;
  const float* x    = (const float*)d_in[0];
  const float* wq   = (const float*)d_in[1];
  const float* wk   = (const float*)d_in[2];
  const float* wv   = (const float*)d_in[3];
  const float* cosT = (const float*)d_in[4];
  const float* sinT = (const float*)d_in[5];

  __bf16* xb = (__bf16*)d_ws;                       // 16384*4096 bf16 (128 MB)
  __bf16* wb = xb + (size_t)MTOT * KDIM;            //  6144*4096 bf16 ( 48 MB)

  // 92,274,688 total elements / 8 per thread / 256 per block = 45056 blocks
  pack_bf16_kernel<<<45056, 256, 0, stream>>>(x, wq, wk, wv, xb, wb);

  dim3 grid(NTOT / 128, MTOT / 128);                // (48, 128)
  qkv_wmma_rope_kernel<<<grid, 256, 0, stream>>>(xb, wb, cosT, sinT,
                                                 (float*)d_out);
}
